// Transformer_69724499083817
// MI455X (gfx1250) — compile-verified, matrix operand order checked by reading, not run on hardware
//
#include <hip/hip_runtime.h>

typedef __attribute__((ext_vector_type(16))) __bf16 v16bf;
typedef __attribute__((ext_vector_type(8)))  float  v8f;

namespace {
constexpr int cB = 64, cS = 256, cD = 128, cH = 8, cDH = 16, cLAB = 200;
constexpr int cNT = cB * cS;
}

// ---------------------------------------------------------------------------
// Fragment loader: CDNA5 16-bit A/B operand layout (16x32 / 32x16), wave32.
// For lane L (m or n = L%16), element e maps to K = 2*(v&3) + (v>=4?16:0) +
// (L>=16?8:0) + (e&1), v=e/2.  K-pairs are contiguous in memory, so this
// reads row-major global data directly; k>=K is zero-filled (padding).
// ---------------------------------------------------------------------------
__device__ __forceinline__ v16bf load_frag(const __bf16* __restrict__ base,
                                           int row, int ld, int k0, int K, bool hi) {
  const __bf16* p = base + (size_t)row * ld;
  v16bf f;
#pragma unroll
  for (int e = 0; e < 16; ++e) {
    int v = e >> 1;
    int k = k0 + 2 * (v & 3) + ((v & 4) ? 16 : 0) + (hi ? 8 : 0) + (e & 1);
    f[e] = (k < K) ? p[k] : (__bf16)0.f;
  }
  return f;
}

// ---------------------------------------------------------------------------
// Generic bf16 GEMM:  out = act(A[MxK] * W[NxK]^T + bias)
// ACT: 0=none 1=relu 2=elu.  EPI: 0=f32 store, 1=bf16 store,
// 2=select-store f32 (etype==tsel ? v : 0), 3=select-accumulate f32.
// One wave32 per 16x16 output tile, K consumed 32 at a time via WMMA.
// ---------------------------------------------------------------------------
template <int ACT, int EPI>
__global__ __launch_bounds__(256) void gemm_bf16(
    const __bf16* __restrict__ A, const __bf16* __restrict__ W,
    const float* __restrict__ bias, float* __restrict__ outF,
    __bf16* __restrict__ outH, int M, int N, int K,
    const int* __restrict__ etype, int tsel) {
  int wave = blockIdx.x * (blockDim.x >> 5) + (threadIdx.x >> 5);
  int lane = threadIdx.x & 31;
  int ntc = N >> 4;
  int tiles = (M >> 4) * ntc;
  if (wave >= tiles) return;
  int mt = wave / ntc, nt = wave % ntc;
  bool hi = lane >= 16;
  int arow = mt * 16 + (lane & 15);
  int wrow = nt * 16 + (lane & 15);
  v8f acc = {};
  for (int k0 = 0; k0 < K; k0 += 32) {
    v16bf a = load_frag(A, arow, K, k0, K, hi);
    v16bf b = load_frag(W, wrow, K, k0, K, hi);
    acc = __builtin_amdgcn_wmma_f32_16x16x32_bf16(false, a, false, b, (short)0,
                                                  acc, false, false);
  }
  int col = nt * 16 + (lane & 15);
  float bv = bias ? bias[col] : 0.f;
#pragma unroll
  for (int j = 0; j < 8; ++j) {
    int row = mt * 16 + j + (hi ? 8 : 0);
    float v = acc[j] + bv;
    if (ACT == 1) v = fmaxf(v, 0.f);
    if (ACT == 2) v = (v > 0.f) ? v : (__expf(v) - 1.f);
    size_t idx = (size_t)row * N + col;
    if (EPI == 0) outF[idx] = v;
    else if (EPI == 1) outH[idx] = (__bf16)v;
    else if (EPI == 2) outF[idx] = (etype[row] == tsel) ? v : 0.f;
    else if (EPI == 3) { if (etype[row] == tsel) outF[idx] += v; }
  }
}

// ---------------------------------------------------------------------------
// Fused flash attention: one wave32 per (batch, head, 16-query tile).
// scores = Q*K^T via WMMA (DH=16 zero-padded to K=32), online softmax with
// 16-lane shfl_xor row reductions, P restaged via 512B LDS slice into an
// A-fragment, O += P*V via a second WMMA.  Avoids the 134MB fp32 score tensor.
// ---------------------------------------------------------------------------
__global__ __launch_bounds__(256) void attn_kernel(const __bf16* __restrict__ qkv,
                                                   const int* __restrict__ pad,
                                                   __bf16* __restrict__ o) {
  __shared__ __bf16 ldsP[8 * 256];
  int wave = blockIdx.x * 8 + (threadIdx.x >> 5);
  int lane = threadIdx.x & 31;
  bool hi = lane >= 16;
  int qt = wave & 15;
  int h = (wave >> 4) & 7;
  int b = wave >> 7;
  __bf16* pl = ldsP + (threadIdx.x >> 5) * 256;

  v16bf qa = load_frag(qkv + h * cDH, b * cS + qt * 16 + (lane & 15), 3 * cD, 0, cDH, hi);
  v8f acc = {};
  float mrun[8], lrun[8];
#pragma unroll
  for (int j = 0; j < 8; ++j) { mrun[j] = -1e30f; lrun[j] = 0.f; }

  for (int kt = 0; kt < cS / 16; ++kt) {
    v16bf kb = load_frag(qkv + cD + h * cDH, b * cS + kt * 16 + (lane & 15), 3 * cD, 0, cDH, hi);
    v8f s = {};
    s = __builtin_amdgcn_wmma_f32_16x16x32_bf16(false, qa, false, kb, (short)0, s, false, false);
    int keycol = kt * 16 + (lane & 15);
    bool masked = pad[b * cS + keycol] != 0;
#pragma unroll
    for (int j = 0; j < 8; ++j) {
      float sv = masked ? -1e9f : s[j] * 0.25f;  // 1/sqrt(16), then key mask
      float rmax = sv;
#pragma unroll
      for (int d2 = 1; d2 < 16; d2 <<= 1) rmax = fmaxf(rmax, __shfl_xor(rmax, d2, 32));
      float mnew = fmaxf(mrun[j], rmax);
      float corr = __expf(mrun[j] - mnew);
      float p = __expf(sv - mnew);
      float rsum = p;
#pragma unroll
      for (int d2 = 1; d2 < 16; d2 <<= 1) rsum += __shfl_xor(rsum, d2, 32);
      lrun[j] = lrun[j] * corr + rsum;
      acc[j] = acc[j] * corr;
      mrun[j] = mnew;
      pl[(j + (hi ? 8 : 0)) * 16 + (lane & 15)] = (__bf16)p;  // C-layout -> LDS
    }
    // P as A-fragment from this wave's private LDS slice (in-order DS per wave)
    v16bf pa;
#pragma unroll
    for (int e = 0; e < 16; ++e) {
      int v = e >> 1;
      int k = 2 * (v & 3) + ((v & 4) ? 16 : 0) + (hi ? 8 : 0) + (e & 1);
      pa[e] = (k < 16) ? pl[(lane & 15) * 16 + k] : (__bf16)0.f;
    }
    // V tile as B-fragment straight from global (n = head dim, k = key index)
    v16bf vb;
#pragma unroll
    for (int e = 0; e < 16; ++e) {
      int v = e >> 1;
      int k = 2 * (v & 3) + ((v & 4) ? 16 : 0) + (hi ? 8 : 0) + (e & 1);
      vb[e] = (k < 16)
                  ? qkv[((size_t)(b * cS + kt * 16 + k)) * (3 * cD) + 2 * cD + h * cDH + (lane & 15)]
                  : (__bf16)0.f;
    }
    acc = __builtin_amdgcn_wmma_f32_16x16x32_bf16(false, pa, false, vb, (short)0, acc, false, false);
  }
#pragma unroll
  for (int j = 0; j < 8; ++j) {
    int row = j + (hi ? 8 : 0);
    int tok = b * cS + qt * 16 + row;
    o[(size_t)tok * cD + h * cDH + (lane & 15)] = (__bf16)(acc[j] / lrun[j]);
  }
}

// ---------------------------------------------------------------------------
// Embedding-table gather-sums (tables are L2-resident: <=10MB each).
// ---------------------------------------------------------------------------
__global__ __launch_bounds__(128) void gather_kernel(
    const float* __restrict__ Tds, const float* __restrict__ Ticd,
    const float* __restrict__ Tpx, const float* __restrict__ Ttr,
    const float* __restrict__ Tmd, const int* __restrict__ codes,
    const int* __restrict__ codes2, const float* __restrict__ lab,
    __bf16* __restrict__ dsicd, __bf16* __restrict__ gpx,
    __bf16* __restrict__ gtr, __bf16* __restrict__ gmd,
    __bf16* __restrict__ labh) {
  int t = blockIdx.x, d = threadIdx.x;
  float ads = 0.f, apx = 0.f, atr = 0.f, amd = 0.f, aicd = 0.f;
  const int* c1 = codes + (size_t)t * 16;
  const int* c2 = codes2 + (size_t)t * 16;
#pragma unroll 4
  for (int k = 0; k < 16; ++k) {
    int i1 = c1[k], i2 = c2[k];
    ads += Tds[(size_t)i1 * 128 + d];
    apx += Tpx[(size_t)i1 * 128 + d];
    atr += Ttr[(size_t)i1 * 128 + d];
    amd += Tmd[(size_t)i1 * 128 + d];
    aicd += Ticd[(size_t)i2 * 128 + d];
  }
  dsicd[(size_t)t * 256 + d] = (__bf16)ads;
  dsicd[(size_t)t * 256 + 128 + d] = (__bf16)aicd;
  gpx[(size_t)t * 128 + d] = (__bf16)apx;
  gtr[(size_t)t * 128 + d] = (__bf16)atr;
  gmd[(size_t)t * 128 + d] = (__bf16)amd;
  for (int d2 = d; d2 < cLAB; d2 += 128)
    labh[(size_t)t * cLAB + d2] = (__bf16)lab[(size_t)t * cLAB + d2];
}

// Residual + LayerNorm in fp32 (writes fp32 state + bf16 GEMM copy).
__global__ __launch_bounds__(128) void resid_ln(const float* __restrict__ xin,
                                                const float* __restrict__ r,
                                                const float* __restrict__ g,
                                                const float* __restrict__ bt,
                                                float* __restrict__ xf,
                                                __bf16* __restrict__ xh) {
  __shared__ float red[128];
  int t = blockIdx.x, d = threadIdx.x;
  size_t idx = (size_t)t * 128 + d;
  float v = xin[idx] + (r ? r[idx] : 0.f);
  red[d] = v; __syncthreads();
  for (int st = 64; st > 0; st >>= 1) { if (d < st) red[d] += red[d + st]; __syncthreads(); }
  float mean = red[0] * (1.f / 128.f);
  __syncthreads();
  float c = v - mean;
  red[d] = c * c; __syncthreads();
  for (int st = 64; st > 0; st >>= 1) { if (d < st) red[d] += red[d + st]; __syncthreads(); }
  float var = red[0] * (1.f / 128.f);
  float o = c * rsqrtf(var + 1e-5f) * g[d] + bt[d];
  xf[idx] = o;
  if (xh) xh[idx] = (__bf16)o;
}

__global__ __launch_bounds__(128) void pe_kernel(__bf16* __restrict__ pes) {
  int s = blockIdx.x, y = threadIdx.x;
  float expo = (float)((y & 1) ? (y - 1) : y) / 128.f;
  float ang = (float)s / powf(10000.f, expo);
  float v = (y & 1) ? cosf(ang) : sinf(ang);
  pes[(size_t)s * 128 + y] = (__bf16)v;
}

__global__ __launch_bounds__(128) void emb_fin(const float* __restrict__ emb,
                                               const float* __restrict__ pelin,
                                               const int* __restrict__ pad,
                                               const float* __restrict__ defv,
                                               float* __restrict__ xf,
                                               __bf16* __restrict__ xh) {
  int t = blockIdx.x, d = threadIdx.x;
  int s = t & 255;
  float v = pad[t] ? defv[d] : emb[(size_t)t * 128 + d];
  v += pelin[(size_t)s * 128 + d];
  xf[(size_t)t * 128 + d] = v;
  xh[(size_t)t * 128 + d] = (__bf16)v;
}

// Final readout: tiny (64x131x128 MACs), keep fp32 scalar for exactness.
__global__ __launch_bounds__(128) void readout_kernel(
    const float* __restrict__ xf, const float* __restrict__ patientF,
    const float* __restrict__ pw1, const float* __restrict__ pb1,
    const float* __restrict__ pw2, const float* __restrict__ pb2,
    const float* __restrict__ rw, const float* __restrict__ rb,
    float* __restrict__ out) {
  int b = blockIdx.x, j = threadIdx.x;
  float h8[8];
#pragma unroll
  for (int i = 0; i < 8; ++i) {
    float a = pb1[i];
    for (int q = 0; q < 3; ++q) a += pw1[i * 3 + q] * patientF[b * 3 + q];
    h8[i] = (a > 0.f) ? a : (__expf(a) - 1.f);
  }
  float pat[3];
#pragma unroll
  for (int tt = 0; tt < 3; ++tt) {
    float a = pb2[tt];
    for (int i = 0; i < 8; ++i) a += pw2[tt * 8 + i] * h8[i];
    pat[tt] = a;
  }
  const float* rep = xf + ((size_t)b * cS + (cS - 1)) * 128;
  float a = rb[j];
  for (int k = 0; k < 128; ++k) a += rw[j * 131 + k] * rep[k];
#pragma unroll
  for (int tt = 0; tt < 3; ++tt) a += rw[j * 131 + 128 + tt] * pat[tt];
  out[(size_t)b * 128 + j] = a;
}

__global__ void f2bf(const float* __restrict__ in, __bf16* __restrict__ out, int n) {
  int i = blockIdx.x * blockDim.x + threadIdx.x;
  if (i < n) out[i] = (__bf16)in[i];
}

// ---------------------------------------------------------------------------
// Host orchestration.  Input index map (setup_inputs + make_params order):
// 0 patientF 1 lab_feat 2 event_type 3 codes 4 codes2 5 pad_mask,
// 6..10 vocab tables, 11 default, 12.. weights as laid out below.
// ---------------------------------------------------------------------------
extern "C" void kernel_launch(void* const* d_in, const int* in_sizes, int n_in,
                              void* d_out, int out_size, void* d_ws, size_t ws_size,
                              hipStream_t stream) {
  (void)in_sizes; (void)n_in; (void)out_size; (void)ws_size;
  auto F = [&](int i) { return (const float*)d_in[i]; };
  const float* patientF = F(0);
  const float* lab_feat = F(1);
  const int* event_type = (const int*)d_in[2];
  const int* codes = (const int*)d_in[3];
  const int* codes2 = (const int*)d_in[4];
  const int* pad = (const int*)d_in[5];

  size_t off = 0;
  auto allocB = [&](size_t bytes) -> void* {
    void* p = (void*)((char*)d_ws + off);
    off += (bytes + 255) & ~(size_t)255;
    return p;
  };
  auto allocH = [&](size_t elems) { return (__bf16*)allocB(elems * sizeof(__bf16)); };
  auto allocF = [&](size_t elems) { return (float*)allocB(elems * sizeof(float)); };
  auto conv = [&](int idx, int n) -> __bf16* {
    __bf16* dst = allocH((size_t)n);
    f2bf<<<(n + 255) / 256, 256, 0, stream>>>(F(idx), dst, n);
    return dst;
  };

  // weights -> bf16 (once per launch; deterministic)
  __bf16* w_diag1 = conv(12, 128 * 256);
  __bf16* w_diag2 = conv(14, 128 * 128);
  __bf16* w_lab1 = conv(16, 128 * cLAB);
  __bf16* w_lab2 = conv(18, 128 * 128);
  __bf16* w_px1 = conv(20, 128 * 128);
  __bf16* w_px2 = conv(22, 128 * 128);
  __bf16* w_tr1 = conv(24, 128 * 128);
  __bf16* w_tr2 = conv(26, 128 * 128);
  __bf16* w_md1 = conv(28, 128 * 128);
  __bf16* w_md2 = conv(30, 128 * 128);
  __bf16 *w_in[3], *w_out[3], *w_f1[3], *w_f2[3];
  for (int l = 0; l < 3; ++l) {
    int base = 32 + 12 * l;
    w_in[l] = conv(base + 0, 384 * 128);
    w_out[l] = conv(base + 2, 128 * 128);
    w_f1[l] = conv(base + 8, 64 * 128);
    w_f2[l] = conv(base + 10, 128 * 64);
  }
  __bf16* w_pe = conv(70, 128 * 128);

  // activations
  __bf16* dsicd = allocH((size_t)cNT * 256);
  __bf16* gpx = allocH((size_t)cNT * 128);
  __bf16* gtr = allocH((size_t)cNT * 128);
  __bf16* gmd = allocH((size_t)cNT * 128);
  __bf16* labh = allocH((size_t)cNT * cLAB);
  __bf16* hbuf = allocH((size_t)cNT * 128);
  float* emb = allocF((size_t)cNT * 128);
  __bf16* pes = allocH((size_t)cS * 128);
  float* pelin = allocF((size_t)cS * 128);
  float* xf = allocF((size_t)cNT * 128);
  __bf16* xh = allocH((size_t)cNT * 128);
  __bf16* qkvb = allocH((size_t)cNT * 384);
  __bf16* ob = allocH((size_t)cNT * 128);
  float* projf = allocF((size_t)cNT * 128);
  __bf16* ffh = allocH((size_t)cNT * 64);
  float* ffo = allocF((size_t)cNT * 128);

  gather_kernel<<<cNT, 128, 0, stream>>>(F(6), F(7), F(8), F(9), F(10), codes, codes2,
                                         lab_feat, dsicd, gpx, gtr, gmd, labh);
  pe_kernel<<<cS, 128, 0, stream>>>(pes);

#define GEMM(ACT, EPI, A, W, BI, OF, OH, M, N, K, ET, TS)                                 \
  do {                                                                                    \
    int _tiles = ((M) / 16) * ((N) / 16);                                                 \
    gemm_bf16<ACT, EPI><<<(_tiles + 7) / 8, 256, 0, stream>>>((A), (W), (BI), (OF), (OH), \
                                                              (M), (N), (K), (ET), (TS)); \
  } while (0)

  GEMM(0, 0, pes, w_pe, F(71), pelin, (__bf16*)nullptr, cS, 128, 128, (const int*)nullptr, 0);

  // branch MLPs, one-hot select fused into the second GEMM's epilogue
  GEMM(2, 1, dsicd, w_diag1, F(13), (float*)nullptr, hbuf, cNT, 128, 256, (const int*)nullptr, 0);
  GEMM(0, 2, hbuf, w_diag2, F(15), emb, (__bf16*)nullptr, cNT, 128, 128, event_type, 0);
  GEMM(2, 1, labh, w_lab1, F(17), (float*)nullptr, hbuf, cNT, 128, cLAB, (const int*)nullptr, 0);
  GEMM(0, 3, hbuf, w_lab2, F(19), emb, (__bf16*)nullptr, cNT, 128, 128, event_type, 1);
  GEMM(2, 1, gpx, w_px1, F(21), (float*)nullptr, hbuf, cNT, 128, 128, (const int*)nullptr, 0);
  GEMM(0, 3, hbuf, w_px2, F(23), emb, (__bf16*)nullptr, cNT, 128, 128, event_type, 2);
  GEMM(2, 1, gtr, w_tr1, F(25), (float*)nullptr, hbuf, cNT, 128, 128, (const int*)nullptr, 0);
  GEMM(0, 3, hbuf, w_tr2, F(27), emb, (__bf16*)nullptr, cNT, 128, 128, event_type, 3);
  GEMM(2, 1, gmd, w_md1, F(29), (float*)nullptr, hbuf, cNT, 128, 128, (const int*)nullptr, 0);
  GEMM(0, 3, hbuf, w_md2, F(31), emb, (__bf16*)nullptr, cNT, 128, 128, event_type, 4);

  emb_fin<<<cNT, 128, 0, stream>>>(emb, pelin, pad, F(11), xf, xh);

  for (int l = 0; l < 3; ++l) {
    int base = 32 + 12 * l;
    GEMM(0, 1, xh, w_in[l], F(base + 1), (float*)nullptr, qkvb, cNT, 384, 128, (const int*)nullptr, 0);
    attn_kernel<<<cB * cH * (cS / 16) / 8, 256, 0, stream>>>(qkvb, pad, ob);
    GEMM(0, 0, ob, w_out[l], F(base + 3), projf, (__bf16*)nullptr, cNT, 128, 128, (const int*)nullptr, 0);
    resid_ln<<<cNT, 128, 0, stream>>>(xf, projf, F(base + 4), F(base + 5), xf, xh);
    GEMM(1, 1, xh, w_f1[l], F(base + 9), (float*)nullptr, ffh, cNT, 64, 128, (const int*)nullptr, 0);
    GEMM(0, 0, ffh, w_f2[l], F(base + 11), ffo, (__bf16*)nullptr, cNT, 128, 64, (const int*)nullptr, 0);
    resid_ln<<<cNT, 128, 0, stream>>>(xf, ffo, F(base + 6), F(base + 7), xf, xh);
  }
  resid_ln<<<cNT, 128, 0, stream>>>(xf, (const float*)nullptr, F(68), F(69), xf, xh);
  readout_kernel<<<cB, 128, 0, stream>>>(xf, patientF, F(72), F(73), F(74), F(75), F(76),
                                         F(77), (float*)d_out);
#undef GEMM
}
